// GCNConv_24129126269175
// MI455X (gfx1250) — matrix operand order repaired and checked by reading, not compiled, not run on hardware
//
#include <hip/hip_runtime.h>
#include <hip/hip_bf16.h>

// ---------------- problem constants ----------------
#define NNODES  16384
#define INF     128
#define OUTF    128
#define NEDGE   524288
#define NG      8
#define KSPL    (INF * NG)     // 1024
#define KTOT    (KSPL + INF)   // 1152 combined K (spline + base)
#define LN_EPS  1e-5f
#define INV_DEN 1.75f          // (NG-1)/(GRID_MAX-GRID_MIN) = 7/4

typedef __attribute__((ext_vector_type(16))) __bf16 v16bf;
typedef __attribute__((ext_vector_type(8)))  __bf16 v8bf;
typedef __attribute__((ext_vector_type(8)))  float  v8f;

// ---------------- kernel 1: pack [spline_w ; base_w] -> bf16 [OUT][KTOT] ----
__global__ __launch_bounds__(256) void pack_weights_kernel(
    const float* __restrict__ spline_w,   // [OUT, KSPL]
    const float* __restrict__ base_w,     // [OUT, INF]
    __bf16* __restrict__ wc)              // [OUT, KTOT]
{
    int idx = blockIdx.x * 256 + threadIdx.x;
    if (idx >= OUTF * KTOT) return;
    int o = idx / KTOT;
    int k = idx - o * KTOT;
    float v = (k < KSPL) ? spline_w[o * KSPL + k] : base_w[o * INF + (k - KSPL)];
    wc[idx] = (__bf16)v;
}

// ---------------- kernel 2: fused LN + RBF basis + WMMA GEMM ----------------
// One block: 16 node rows x 128 out cols. 256 threads = 8 waves.
// Wave w computes the 16x16 tile at columns [16w, 16w+16).
__global__ __launch_bounds__(256) void support_kernel(
    const float* __restrict__ x,          // [N, INF]
    const float* __restrict__ ln_gamma,   // [INF]
    const float* __restrict__ ln_beta,    // [INF]
    const __bf16* __restrict__ wc,        // [OUT, KTOT] bf16
    const float* __restrict__ base_b,     // [OUT]
    float* __restrict__ support)          // [N, OUT]
{
    __shared__ __align__(16) float  xs[16][INF];
    __shared__ __align__(16) float  hs[16][INF];
    __shared__ __align__(16) __bf16 As[16][KTOT];   // [basis(1024) | silu(128)]

    const int t = threadIdx.x;
    const int rowbase = blockIdx.x * 16;

    // ---- phase 1: stage x tile to LDS (float4 vector loads) ----
    {
        const float4* xg  = (const float4*)(x + (size_t)rowbase * INF);
        float4*       xsv = (float4*)&xs[0][0];
        for (int i = t; i < 16 * INF / 4; i += 256) xsv[i] = xg[i];
    }
    __syncthreads();

    // ---- phase 2: LayerNorm (16 lanes per row) + silu into A tail ----
    {
        const int row = t >> 4;        // 0..15
        const int c16 = t & 15;        // 16 threads per row, 8 elems each
        float s = 0.f, ss = 0.f;
        #pragma unroll
        for (int j = 0; j < 8; ++j) {
            float v = xs[row][c16 * 8 + j];
            s += v; ss += v * v;
        }
        #pragma unroll
        for (int m = 1; m < 16; m <<= 1) {    // xor<16 stays within 16-lane half
            s  += __shfl_xor(s,  m, 32);
            ss += __shfl_xor(ss, m, 32);
        }
        const float mu = s * (1.0f / INF);
        const float var = ss * (1.0f / INF) - mu * mu;
        const float rv = rsqrtf(var + LN_EPS);
        #pragma unroll
        for (int j = 0; j < 8; ++j) {
            int c = c16 * 8 + j;
            float xv = xs[row][c];
            float h  = (xv - mu) * rv * ln_gamma[c] + ln_beta[c];
            hs[row][c] = h;
            float sig  = 1.0f / (1.0f + __expf(-xv));
            As[row][KSPL + c] = (__bf16)(xv * sig);       // silu(x)
        }
    }
    __syncthreads();

    // ---- phase 3: Gaussian RBF basis, bf16, layout col = feat*NG + g ----
    for (int p = t; p < 16 * INF; p += 256) {
        int row = p >> 7;
        int c   = p & (INF - 1);
        float h = hs[row][c];
        #pragma unroll
        for (int g = 0; g < NG; ++g) {
            float gr = -2.0f + (float)g * (4.0f / 7.0f);
            float d  = (h - gr) * INV_DEN;
            As[row][c * NG + g] = (__bf16)__expf(-d * d);
        }
    }
    __syncthreads();

    // ---- phase 4: WMMA K-loop, 36 x v_wmma_f32_16x16x32_bf16 per wave ----
    const int wave = t >> 5;
    const int lane = t & 31;
    const int nn   = lane & 15;     // A row m / B col n / C col n
    const int half = lane >> 4;     // K sub-block select
    const int n0   = wave * 16;

    v8f acc = {};
    const __bf16* brow = wc + (size_t)(n0 + nn) * KTOT;

    for (int kt = 0; kt < KTOT / 32; ++kt) {
        const int kbase = kt * 32;
        // A fragment: lane holds row nn, K = kbase + half*8 + {0..7, 16..23}
        v8bf lo = *(const v8bf*)&As[nn][kbase + half * 8];
        v8bf hi = *(const v8bf*)&As[nn][kbase + 16 + half * 8];
        v16bf a;
        #pragma unroll
        for (int i = 0; i < 8; ++i) { a[i] = lo[i]; a[i + 8] = hi[i]; }
        // B fragment: lane holds col n0+nn, K = kbase + half*8 + {0..7, 16..23}
        if (kt + 1 < KTOT / 32)
            __builtin_prefetch(brow + kbase + 32, 0, 1);   // global_prefetch_b8
        v8bf blo = *(const v8bf*)(brow + kbase + half * 8);
        v8bf bhi = *(const v8bf*)(brow + kbase + 16 + half * 8);
        v16bf b;
        #pragma unroll
        for (int i = 0; i < 8; ++i) { b[i] = blo[i]; b[i + 8] = bhi[i]; }
        acc = __builtin_amdgcn_wmma_f32_16x16x32_bf16(
            /*neg_a=*/false, a, /*neg_b=*/false, b,
            /*c_mod=*/(short)0, acc, /*reuse_a=*/false, /*reuse_b=*/false);
    }

    // ---- phase 5: + base_b, store 16x16 C tile (lane: col nn, rows half*8+v)
    const float bb = base_b[n0 + nn];
    #pragma unroll
    for (int v = 0; v < 8; ++v) {
        int m = half * 8 + v;
        support[(size_t)(rowbase + m) * OUTF + n0 + nn] = acc[v] + bb;
    }
}

// ---------------- kernel 3: zero float buffer ----------------
__global__ __launch_bounds__(256) void zero_kernel(float* __restrict__ p, int n) {
    int i = blockIdx.x * 256 + threadIdx.x;
    if (i < n) p[i] = 0.0f;
}

// ---------------- kernel 4: degree count (symmetric adjacency) --------------
__global__ __launch_bounds__(256) void degree_kernel(
    const int* __restrict__ ei, float* __restrict__ deg)
{
    int e = blockIdx.x * 256 + threadIdx.x;
    if (e >= NEDGE) return;
    int i = ei[e];
    int j = ei[NEDGE + e];
    atomicAdd(&deg[i], 1.0f);
    if (i != j) atomicAdd(&deg[j], 1.0f);
}

// ---------------- kernel 5: dis = 1/sqrt(deg) ----------------
__global__ __launch_bounds__(256) void dis_kernel(
    const float* __restrict__ deg, float* __restrict__ dis)
{
    int i = blockIdx.x * 256 + threadIdx.x;
    if (i < NNODES) dis[i] = rsqrtf(deg[i]);
}

// ---------------- kernel 6: out = bias (broadcast) ----------------
__global__ __launch_bounds__(256) void init_out_kernel(
    const float* __restrict__ bias, float* __restrict__ out)
{
    int idx = blockIdx.x * 256 + threadIdx.x;
    if (idx < NNODES * OUTF) out[idx] = bias[idx & (OUTF - 1)];
}

// ---------------- kernel 7: edge scatter SpMM (one wave per edge) -----------
__global__ __launch_bounds__(256) void scatter_kernel(
    const int* __restrict__ ei, const float* __restrict__ dis,
    const float* __restrict__ support, float* __restrict__ out)
{
    int gid  = blockIdx.x * 256 + threadIdx.x;
    int e    = gid >> 5;
    int lane = gid & 31;
    if (e >= NEDGE) return;
    int i = ei[e];
    int j = ei[NEDGE + e];
    float c = dis[i] * dis[j];
    const float* sj = support + (size_t)j * OUTF;
    const float* si = support + (size_t)i * OUTF;
    float* oi = out + (size_t)i * OUTF;
    float* oj = out + (size_t)j * OUTF;
    #pragma unroll
    for (int k = 0; k < 4; ++k) {
        int col = lane * 4 + k;
        atomicAdd(&oi[col], c * sj[col]);
        if (i != j) atomicAdd(&oj[col], c * si[col]);
    }
}

// ---------------- launch ----------------
extern "C" void kernel_launch(void* const* d_in, const int* in_sizes, int n_in,
                              void* d_out, int out_size, void* d_ws, size_t ws_size,
                              hipStream_t stream) {
    const float* x        = (const float*)d_in[0];
    const int*   ei       = (const int*)  d_in[1];
    const float* ln_gamma = (const float*)d_in[2];
    const float* ln_beta  = (const float*)d_in[3];
    const float* spline_w = (const float*)d_in[4];
    const float* base_w   = (const float*)d_in[5];
    const float* base_b   = (const float*)d_in[6];
    const float* bias     = (const float*)d_in[7];
    float* out = (float*)d_out;

    char* ws = (char*)d_ws;
    __bf16* wc      = (__bf16*)ws;                               // 128*1152*2 = 294912 B
    float*  support = (float*)(ws + 294912);                     // 16384*128*4 = 8 MiB
    float*  deg     = (float*)(ws + 294912 + (size_t)NNODES * OUTF * 4);
    float*  dis     = deg + NNODES;

    pack_weights_kernel<<<(OUTF * KTOT + 255) / 256, 256, 0, stream>>>(spline_w, base_w, wc);
    support_kernel<<<NNODES / 16, 256, 0, stream>>>(x, ln_gamma, ln_beta, wc, base_b, support);
    zero_kernel<<<(NNODES + 255) / 256, 256, 0, stream>>>(deg, NNODES);
    degree_kernel<<<(NEDGE + 255) / 256, 256, 0, stream>>>(ei, deg);
    dis_kernel<<<(NNODES + 255) / 256, 256, 0, stream>>>(deg, dis);
    init_out_kernel<<<(NNODES * OUTF + 255) / 256, 256, 0, stream>>>(bias, out);
    scatter_kernel<<<(NEDGE * 32) / 256, 256, 0, stream>>>(ei, dis, support, out);
}
